// MaskEntropyModel_85822036509204
// MI455X (gfx1250) — matrix operand order repaired and checked by reading, not compiled, not run on hardware
//
#include <hip/hip_runtime.h>
#include <hip/hip_bf16.h>

typedef __attribute__((ext_vector_type(16))) _Float16 v16h;
typedef __attribute__((ext_vector_type(8)))  _Float16 v8h;
typedef __attribute__((ext_vector_type(8)))  float    v8f;
typedef int v4i_vs __attribute__((vector_size(16)));   // matches builtin param type

#define AS1 __attribute__((address_space(1)))
#define AS3 __attribute__((address_space(3)))

// CDNA5 async memory->LDS path (ASYNCcnt), guarded so we fall back cleanly.
#if defined(__has_builtin)
# if __has_builtin(__builtin_amdgcn_global_load_async_to_lds_b128) && \
     __has_builtin(__builtin_amdgcn_s_wait_asynccnt)
#  define USE_ASYNC_LDS 1
# endif
#endif
#ifndef USE_ASYNC_LDS
# define USE_ASYNC_LDS 0
#endif

#if USE_ASYNC_LDS
#define GASYNC_B128(gp, lp)                                                      \
  __builtin_amdgcn_global_load_async_to_lds_b128(                                \
      (AS1 v4i_vs*)(gp), (AS3 v4i_vs*)(lp), 0, 0)
#endif

// ---------------- model dims (hardcoded from setup_inputs) ----------------
#define BATCH 16
#define CDIM  192
#define LTOK  4096
#define NHW   6
#define HD    32
#define CHH   4
#define CHD   16
#define NCTX  256
#define CC    64
#define HID   768
#define TW    1024          // BATCH * 64 windows
#define MT    65536         // BATCH * LTOK

__device__ __forceinline__ v8h ld8(const _Float16* p) { return *(const v8h*)p; }
__device__ __forceinline__ v16h cat16(v8h lo, v8h hi) {
  return __builtin_shufflevector(lo, hi, 0,1,2,3,4,5,6,7,8,9,10,11,12,13,14,15);
}
#define WMMA(A_, B_, C_) \
  __builtin_amdgcn_wmma_f32_16x16x32_f16(false, (A_), false, (B_), (short)0, (C_), false, false)

// ======================= f32 -> f16 convert ========================
__global__ void k_cvt_f16(const float* __restrict__ src, _Float16* __restrict__ dst, int n) {
  int i = blockIdx.x * blockDim.x + threadIdx.x;
  if (i < n) dst[i] = (_Float16)src[i];
}

// ============ shifted window partition: x(B,H,W,C) -> xw(TW,64,C) f16 ============
__global__ __launch_bounds__(192) void k_win_part(const float* __restrict__ x, _Float16* __restrict__ xw) {
  int t = blockIdx.x;           // global window-token 0..65535
  int c = threadIdx.x;          // channel 0..191
  int win = t >> 6, tok = t & 63;
  int b = win >> 6, wi = win & 63;
  int wh = wi >> 3, ww = wi & 7;
  int r  = tok >> 3, cc = tok & 7;
  int sh = ((wh * 8 + r) + 4) & 63;   // roll(-4)
  int sw = ((ww * 8 + cc) + 4) & 63;
  xw[(size_t)t * CDIM + c] = (_Float16)x[(((size_t)b * LTOK) + sh * 64 + sw) * CDIM + c];
}

// ======================= generic WMMA GEMM =========================
// C[M,N] = A[M,K](f16) * B[K,N](f16) + bias ; act=1 -> exact GELU
// block 128 thr (4 waves), tile 64x64.  Software-pipelined K loop:
//  - A tile: async global->LDS (double-buffered) when available
//  - B tile: global->regs early, transposed reg->LDS scatter at commit
__global__ __launch_bounds__(128) void k_gemm(const _Float16* __restrict__ A,
                                              const _Float16* __restrict__ Bm,
                                              const float* __restrict__ bias,
                                              _Float16* __restrict__ outh,
                                              float* __restrict__ outf,
                                              int M, int N, int K, int act) {
  __shared__ __align__(16) _Float16 sA[2][64 * 32];
  __shared__ __align__(16) _Float16 sBt[64 * 32];   // [n][k] transposed
  int tid = threadIdx.x;
  int wv = tid >> 5, lane = tid & 31;
  int half = lane >> 4, r = lane & 15;
  int m0 = blockIdx.x * 64, n0 = blockIdx.y * 64;

  const int arow0 = tid >> 2, acch = tid & 3;       // A chunks: rows arow0, arow0+32
  const int brow0 = tid >> 3, bcch = tid & 7;       // B chunks: rows brow0, brow0+16

  v8h rB[2];
#if !USE_ASYNC_LDS
  v8h rA[2];
#endif

  auto issueA = [&](int k0, int bufi) {
#pragma unroll
    for (int it = 0; it < 2; it++) {
      int row = arow0 + 32 * it;
      const _Float16* g = &A[(size_t)(m0 + row) * K + k0 + acch * 8];
#if USE_ASYNC_LDS
      GASYNC_B128(g, &sA[bufi][row * 32 + acch * 8]);
#else
      rA[it] = ld8(g);
#endif
    }
  };
  auto loadB = [&](int k0) {
#pragma unroll
    for (int it = 0; it < 2; it++) {
      int row = brow0 + 16 * it;
      rB[it] = ld8(&Bm[(size_t)(k0 + row) * N + n0 + bcch * 8]);
    }
  };

  issueA(0, 0);
  loadB(0);
  v8f acc[4] = {};
  int buf = 0;
  for (int k0 = 0; k0 < K; k0 += 32) {
    // ---- commit staged tile ----
#pragma unroll
    for (int it = 0; it < 2; it++) {                 // B transpose scatter
      int row = brow0 + 16 * it;
#pragma unroll
      for (int jj = 0; jj < 8; jj++) sBt[(bcch * 8 + jj) * 32 + row] = rB[it][jj];
    }
#if USE_ASYNC_LDS
    __builtin_amdgcn_s_wait_asynccnt(0);             // A tile landed in LDS
#else
#pragma unroll
    for (int it = 0; it < 2; it++)
      *(v8h*)&sA[buf][(arow0 + 32 * it) * 32 + acch * 8] = rA[it];
#endif
    __syncthreads();
    if (k0 + 32 < K) { issueA(k0 + 32, buf ^ 1); loadB(k0 + 32); }  // prefetch next
    // ---- compute ----
    const _Float16* ap = &sA[buf][(wv * 16 + r) * 32 + half * 8];
    v16h a = cat16(ld8(ap), ld8(ap + 16));
#pragma unroll
    for (int tc = 0; tc < 4; tc++) {
      const _Float16* bp = &sBt[(tc * 16 + r) * 32 + half * 8];
      v16h b = cat16(ld8(bp), ld8(bp + 16));
      acc[tc] = WMMA(a, b, acc[tc]);
    }
    __syncthreads();
    buf ^= 1;
  }
#pragma unroll
  for (int tc = 0; tc < 4; tc++) {
#pragma unroll
    for (int j = 0; j < 8; j++) {                    // C layout: M=j+8*half, N=lane%16
      int m = m0 + wv * 16 + j + 8 * half;
      int n = n0 + tc * 16 + r;
      float v = acc[tc][j];
      if (bias) v += bias[n];
      if (act)  v = 0.5f * v * (1.0f + erff(v * 0.70710678118654752f));
      size_t idx = (size_t)m * N + n;
      if (outf) outf[idx] = v;
      if (outh) outh[idx] = (_Float16)v;
    }
  }
}

// ================ fused shifted-window attention ==================
// 1 block = 1 window (1024 blocks), 6 waves = 6 heads.  All LDS is wave-private.
__global__ __launch_bounds__(192) void k_winattn(const _Float16* __restrict__ qkv,
                                                 const float* __restrict__ rpb,
                                                 _Float16* __restrict__ outp) {
  __shared__ __align__(16) _Float16 sP[NHW][16][64];    // P slab, reused per row-tile
  __shared__ __align__(16) _Float16 sVt[NHW][HD][64];   // V transposed [ch][tok]
  int win = blockIdx.x;
  int wi = win & 63, wh = wi >> 3, ww = wi & 7;
  int head = threadIdx.x >> 5;
  int lane = threadIdx.x & 31;
  int half = lane >> 4, r = lane & 15;
  const _Float16* base = qkv + (size_t)win * 64 * 576;

  // ---- stage V^T for this head (coalesced b128 reads, b16 scatter) ----
#pragma unroll
  for (int t2 = 0; t2 < 2; t2++) {
    int tok = lane + t2 * 32;
    const _Float16* vp = base + (size_t)tok * 576 + 2 * CDIM + head * HD;
    v8h p0 = ld8(vp), p1 = ld8(vp + 8), p2 = ld8(vp + 16), p3 = ld8(vp + 24);
#pragma unroll
    for (int c = 0; c < 8; c++) {
      sVt[head][c][tok]      = p0[c];
      sVt[head][8 + c][tok]  = p1[c];
      sVt[head][16 + c][tok] = p2[c];
      sVt[head][24 + c][tok] = p3[c];
    }
  }

  // ---- scores S = Q K^T (4x4 tiles, K-dim = 32); fragments via b128 loads ----
  v8f s[4][4] = {};
#pragma unroll
  for (int tr = 0; tr < 4; tr++) {
    const _Float16* ap = base + (size_t)(tr * 16 + r) * 576 + head * HD + half * 8;
    v16h a = cat16(ld8(ap), ld8(ap + 16));
#pragma unroll
    for (int tc = 0; tc < 4; tc++) {
      const _Float16* bp = base + (size_t)(tc * 16 + r) * 576 + CDIM + head * HD + half * 8;
      v16h bk = cat16(ld8(bp), ld8(bp + 16));
      s[tr][tc] = WMMA(a, bk, s[tr][tc]);
    }
  }
  // ---- scale + relative-position bias + shift mask ----
  const float scale = 0.17677669529663687f;          // 1/sqrt(32)
#pragma unroll
  for (int tr = 0; tr < 4; tr++)
#pragma unroll
    for (int tc = 0; tc < 4; tc++)
#pragma unroll
      for (int j = 0; j < 8; j++) {
        int n = tr * 16 + j + 8 * half;              // query token
        int m = tc * 16 + r;                         // key token
        int ri = n >> 3, ci = n & 7, rj = m >> 3, cj = m & 7;
        int rpi = (ri - rj + 7) * 15 + (ci - cj + 7);
        float bias = rpb[rpi * NHW + head];
        int gnh = wh * 8 + ri, gnw = ww * 8 + ci;
        int gmh = wh * 8 + rj, gmw = ww * 8 + cj;
        int cn = (gnh < 56 ? 0 : (gnh < 60 ? 1 : 2)) * 3 + (gnw < 56 ? 0 : (gnw < 60 ? 1 : 2));
        int cm = (gmh < 56 ? 0 : (gmh < 60 ? 1 : 2)) * 3 + (gmw < 56 ? 0 : (gmw < 60 ? 1 : 2));
        s[tr][tc][j] = s[tr][tc][j] * scale + bias + ((cn != cm) ? -100.0f : 0.0f);
      }
  // ---- per row-tile: softmax -> P slab -> O = P V ----
#pragma unroll
  for (int tr = 0; tr < 4; tr++) {
#pragma unroll
    for (int j = 0; j < 8; j++) {
      float mx = -3.0e38f;
#pragma unroll
      for (int tc = 0; tc < 4; tc++) mx = fmaxf(mx, s[tr][tc][j]);
      for (int d = 1; d < 16; d <<= 1) mx = fmaxf(mx, __shfl_xor(mx, d, 32));
      float sum = 0.f;
#pragma unroll
      for (int tc = 0; tc < 4; tc++) { float e = __expf(s[tr][tc][j] - mx); s[tr][tc][j] = e; sum += e; }
      for (int d = 1; d < 16; d <<= 1) sum += __shfl_xor(sum, d, 32);
      float inv = 1.0f / sum;
      int row = j + 8 * half;
#pragma unroll
      for (int tc = 0; tc < 4; tc++)
        sP[head][row][tc * 16 + r] = (_Float16)(s[tr][tc][j] * inv);
    }
    v8f z = {};
    v8f o0 = z, o1 = z;
#pragma unroll
    for (int kc = 0; kc < 2; kc++) {
      const _Float16* pp = &sP[head][r][kc * 32 + half * 8];
      v16h pa = cat16(ld8(pp), ld8(pp + 16));
      const _Float16* v0p = &sVt[head][r][kc * 32 + half * 8];
      v16h vb0 = cat16(ld8(v0p), ld8(v0p + 16));
      o0 = WMMA(pa, vb0, o0);
      const _Float16* v1p = &sVt[head][16 + r][kc * 32 + half * 8];
      v16h vb1 = cat16(ld8(v1p), ld8(v1p + 16));
      o1 = WMMA(pa, vb1, o1);
    }
#pragma unroll
    for (int j = 0; j < 8; j++) {
      int m = tr * 16 + j + 8 * half;
      size_t ob = ((size_t)win * 64 + m) * CDIM + head * HD;
      outp[ob + r]      = (_Float16)o0[j];
      outp[ob + 16 + r] = (_Float16)o1[j];
    }
  }
}

// ===================== fused cross attention ======================
// grid (L/64, CH, B), 4 waves; each wave: 16 queries vs 256 keys, chd=16 padded to K=32.
__global__ __launch_bounds__(128) void k_crossattn(const _Float16* __restrict__ qc,
                                                   const _Float16* __restrict__ kv,
                                                   _Float16* __restrict__ co) {
  __shared__ __align__(16) _Float16 sP[4][16][256];
  __shared__ __align__(16) _Float16 sVt[CHD][NCTX];     // V transposed [ch][tok]
  int qt = blockIdx.x, h = blockIdx.y, b = blockIdx.z;
  int wv = threadIdx.x >> 5, lane = threadIdx.x & 31;
  int half = lane >> 4, r = lane & 15;
  int q0 = qt * 64 + wv * 16;
  const size_t kvb = (size_t)b * NCTX * (2 * CC);

  // ---- stage V^T cooperatively (whole block shares (b,h)) ----
#pragma unroll
  for (int t2 = 0; t2 < 2; t2++) {
    int tok = threadIdx.x + t2 * 128;
    const _Float16* vp = kv + kvb + (size_t)tok * (2 * CC) + CC + h * CHD;
    v8h p0 = ld8(vp), p1 = ld8(vp + 8);
#pragma unroll
    for (int c = 0; c < 8; c++) { sVt[c][tok] = p0[c]; sVt[8 + c][tok] = p1[c]; }
  }

  // ---- Q fragment (K 16 -> 32 zero-padded in upper half) ----
  const _Float16* qp = qc + ((size_t)b * LTOK + q0) * CC + h * CHD + (size_t)r * CC + half * 8;
  v8h z8 = {};
  v16h a = cat16(ld8(qp), z8);

  v8f s[16] = {};
#pragma unroll
  for (int tc = 0; tc < 16; tc++) {
    v16h bk = {};
    {                                                 // elements s >= 8 (K rows >= 16) are the zero pad
      const _Float16* kp = kv + kvb + (size_t)(tc * 16 + r) * (2 * CC) + h * CHD + half * 8;
      bk = cat16(ld8(kp), z8);
    }
    s[tc] = WMMA(a, bk, s[tc]);
  }
  // ---- softmax over 256 keys, scale = chd^-0.5 ----
#pragma unroll
  for (int j = 0; j < 8; j++) {
    float mx = -3.0e38f;
#pragma unroll
    for (int tc = 0; tc < 16; tc++) { s[tc][j] *= 0.25f; mx = fmaxf(mx, s[tc][j]); }
    for (int d = 1; d < 16; d <<= 1) mx = fmaxf(mx, __shfl_xor(mx, d, 32));
    float sum = 0.f;
#pragma unroll
    for (int tc = 0; tc < 16; tc++) { float e = __expf(s[tc][j] - mx); s[tc][j] = e; sum += e; }
    for (int d = 1; d < 16; d <<= 1) sum += __shfl_xor(sum, d, 32);
    float inv = 1.0f / sum;
    int row = j + 8 * half;
#pragma unroll
    for (int tc = 0; tc < 16; tc++)
      sP[wv][row][tc * 16 + r] = (_Float16)(s[tc][j] * inv);
  }
  __syncthreads();
  // ---- O = P V  (K=256, N=16) ----
  v8f o = {};
#pragma unroll
  for (int kc = 0; kc < 8; kc++) {
    const _Float16* pp = &sP[wv][r][kc * 32 + half * 8];
    v16h pa = cat16(ld8(pp), ld8(pp + 16));
    const _Float16* vp2 = &sVt[r][kc * 32 + half * 8];
    v16h vb = cat16(ld8(vp2), ld8(vp2 + 16));
    o = WMMA(pa, vb, o);
  }
#pragma unroll
  for (int j = 0; j < 8; j++) {
    int q = q0 + j + 8 * half;
    co[((size_t)b * LTOK + q) * CC + h * CHD + r] = (_Float16)o[j];
  }
}

// ================ LayerNorm(y)+residual (optional window-reverse gather) ================
__global__ __launch_bounds__(192) void k_ln_add(const float* __restrict__ y,
                                                const float* __restrict__ res,
                                                const float* __restrict__ g,
                                                const float* __restrict__ bb,
                                                float* __restrict__ outf,
                                                _Float16* __restrict__ outh,
                                                int gather) {
  __shared__ float red[192];
  __shared__ float s_m, s_r;
  int t = blockIdx.x, c = threadIdx.x;
  size_t src;
  if (gather) {                                       // window reverse + roll(+4)
    int b = t >> 12, l = t & 4095, hh = l >> 6, wp = l & 63;
    int hs = (hh - 4) & 63, ws = (wp - 4) & 63;
    int wi = (hs >> 3) * 8 + (ws >> 3);
    int tok = (hs & 7) * 8 + (ws & 7);
    src = (((size_t)b * 64 + wi) * 64 + tok) * CDIM + c;
  } else {
    src = (size_t)t * CDIM + c;
  }
  float v = y[src];
  red[c] = v; __syncthreads();
  for (int s = 96; s >= 6; s >>= 1) { if (c < s) red[c] += red[c + s]; __syncthreads(); }
  if (c == 0) { float m = red[0]+red[1]+red[2]+red[3]+red[4]+red[5]; s_m = m * (1.0f/192.0f); }
  __syncthreads();
  float d = v - s_m;
  red[c] = d * d; __syncthreads();
  for (int s = 96; s >= 6; s >>= 1) { if (c < s) red[c] += red[c + s]; __syncthreads(); }
  if (c == 0) { float vv = red[0]+red[1]+red[2]+red[3]+red[4]+red[5];
                s_r = rsqrtf(vv * (1.0f/192.0f) + 1e-5f); }
  __syncthreads();
  size_t oidx = (size_t)t * CDIM + c;
  float o = res[oidx] + d * s_r * g[c] + bb[c];
  outf[oidx] = o;
  if (outh) outh[oidx] = (_Float16)o;
}

// ============================ driver ==============================
extern "C" void kernel_launch(void* const* d_in, const int* in_sizes, int n_in,
                              void* d_out, int out_size, void* d_ws, size_t ws_size,
                              hipStream_t stream) {
  (void)in_sizes; (void)n_in; (void)out_size; (void)ws_size;
  const float* x     = (const float*)d_in[0];
  const float* ctx   = (const float*)d_in[1];
  const float* qkv_w = (const float*)d_in[2];  const float* qkv_b = (const float*)d_in[3];
  const float* prj_w = (const float*)d_in[4];  const float* prj_b = (const float*)d_in[5];
  const float* rpb   = (const float*)d_in[6];
  const float* n1g   = (const float*)d_in[7];  const float* n1b   = (const float*)d_in[8];
  const float* q_w   = (const float*)d_in[9];  const float* q_b   = (const float*)d_in[10];
  const float* kv_w  = (const float*)d_in[11]; const float* kv_b  = (const float*)d_in[12];
  const float* cp_w  = (const float*)d_in[13]; const float* cp_b  = (const float*)d_in[14];
  const float* cng   = (const float*)d_in[15]; const float* cnb   = (const float*)d_in[16];
  const float* n2g   = (const float*)d_in[17]; const float* n2b   = (const float*)d_in[18];
  const float* f1_w  = (const float*)d_in[19]; const float* f1_b  = (const float*)d_in[20];
  const float* f2_w  = (const float*)d_in[21]; const float* f2_b  = (const float*)d_in[22];
  float* out = (float*)d_out;

  // ---- workspace layout (256B aligned) ----
  char* w = (char*)d_ws;
  size_t off = 0;
  auto take = [&](size_t bytes) { size_t o = off; off += (bytes + 255) & ~(size_t)255; return o; };
  _Float16* wqkv = (_Float16*)(w + take((size_t)CDIM * 3 * CDIM * 2));
  _Float16* wprj = (_Float16*)(w + take((size_t)CDIM * CDIM * 2));
  _Float16* wq   = (_Float16*)(w + take((size_t)CDIM * CC * 2));
  _Float16* wkv  = (_Float16*)(w + take((size_t)CC * 2 * CC * 2));
  _Float16* wcp  = (_Float16*)(w + take((size_t)CC * CDIM * 2));
  _Float16* wf1  = (_Float16*)(w + take((size_t)CDIM * HID * 2));
  _Float16* wf2  = (_Float16*)(w + take((size_t)HID * CDIM * 2));
  _Float16* ctxh = (_Float16*)(w + take((size_t)BATCH * NCTX * CC * 2));
  _Float16* AH   = (_Float16*)(w + take((size_t)MT * CDIM * 2));   // xw -> attnout -> x1h -> x2h
  _Float16* BIGH = (_Float16*)(w + take((size_t)MT * HID * 2));    // qkv f16 -> mlp hidden
  _Float16* QCH  = (_Float16*)(w + take((size_t)MT * CC * 2));
  _Float16* KVH  = (_Float16*)(w + take((size_t)BATCH * NCTX * 2 * CC * 2));
  _Float16* COH  = (_Float16*)(w + take((size_t)MT * CC * 2));
  float*    F32A = (float*)(w + take((size_t)MT * CDIM * 4));      // proj -> cproj -> fc2 out
  float*    X12  = (float*)(w + take((size_t)MT * CDIM * 4));      // residual stream

  auto cvt = [&](const float* s, _Float16* dptr, int n) {
    k_cvt_f16<<<(n + 255) / 256, 256, 0, stream>>>(s, dptr, n);
  };
  // 0) weights + context to f16
  cvt(qkv_w, wqkv, CDIM * 3 * CDIM);
  cvt(prj_w, wprj, CDIM * CDIM);
  cvt(q_w,   wq,   CDIM * CC);
  cvt(kv_w,  wkv,  CC * 2 * CC);
  cvt(cp_w,  wcp,  CC * CDIM);
  cvt(f1_w,  wf1,  CDIM * HID);
  cvt(f2_w,  wf2,  HID * CDIM);
  cvt(ctx,   ctxh, BATCH * NCTX * CC);

  // 1) shifted window partition
  k_win_part<<<MT, 192, 0, stream>>>(x, AH);
  // 2) QKV projection: (MT,192)x(192,576)+b -> f16
  k_gemm<<<dim3(MT / 64, (3 * CDIM) / 64), 128, 0, stream>>>(AH, wqkv, qkv_b, BIGH, nullptr,
                                                             MT, 3 * CDIM, CDIM, 0);
  // 3) fused window attention -> attnout f16 (reuses AH)
  k_winattn<<<TW, 192, 0, stream>>>(BIGH, rpb, AH);
  // 4) output projection -> f32
  k_gemm<<<dim3(MT / 64, CDIM / 64), 128, 0, stream>>>(AH, wprj, prj_b, nullptr, F32A,
                                                       MT, CDIM, CDIM, 0);
  // 5) x1 = x + LN(window_reverse(proj)) ; also f16 copy into AH
  k_ln_add<<<MT, 192, 0, stream>>>(F32A, x, n1g, n1b, X12, AH, 1);
  // 6) cross-attn Q projection
  k_gemm<<<dim3(MT / 64, CC / 64), 128, 0, stream>>>(AH, wq, q_b, QCH, nullptr,
                                                     MT, CC, CDIM, 0);
  // 7) KV projection on context
  k_gemm<<<dim3((BATCH * NCTX) / 64, (2 * CC) / 64), 128, 0, stream>>>(ctxh, wkv, kv_b, KVH, nullptr,
                                                                      BATCH * NCTX, 2 * CC, CC, 0);
  // 8) fused cross attention
  k_crossattn<<<dim3(LTOK / 64, CHH, BATCH), 128, 0, stream>>>(QCH, KVH, COH);
  // 9) cross output projection -> f32
  k_gemm<<<dim3(MT / 64, CDIM / 64), 128, 0, stream>>>(COH, wcp, cp_b, nullptr, F32A,
                                                       MT, CDIM, CC, 0);
  // 10) x2 = x1 + LN(cproj) (in-place residual) ; f16 copy into AH
  k_ln_add<<<MT, 192, 0, stream>>>(F32A, X12, cng, cnb, X12, AH, 0);
  // 11) MLP fc1 + exact GELU -> f16 hidden
  k_gemm<<<dim3(MT / 64, HID / 64), 128, 0, stream>>>(AH, wf1, f1_b, BIGH, nullptr,
                                                      MT, HID, CDIM, 1);
  // 12) MLP fc2 -> f32
  k_gemm<<<dim3(MT / 64, CDIM / 64), 128, 0, stream>>>(BIGH, wf2, f2_b, nullptr, F32A,
                                                       MT, CDIM, HID, 0);
  // 13) out = x2 + LN(fc2)
  k_ln_add<<<MT, 192, 0, stream>>>(F32A, X12, n2g, n2b, out, nullptr, 0);
}
